// WidthFormer_91018946937227
// MI455X (gfx1250) — matrix operand order repaired and checked by reading, not compiled, not run on hardware
//
#include <hip/hip_runtime.h>
#include <math.h>

// ---------------------------------------------------------------------------
// WidthFormer block for MI455X (gfx1250, wave32, WMMA).
// GEMMs use v_wmma_f32_16x16x32_f16 (f16 in, f32 accumulate).
// Weights are pre-swizzled into WMMA B-fragment order so every B fragment is
// two b128 loads (previous version gathered 16x u16 per fragment).
// A fragments are two b128 loads (exact per the 16-bit A register layout).
// Cross-attention is fused per (b,w) column: kn/vn/k/v stay in LDS
// (saves ~550MB HBM round-trip; mandatory traffic ~150MB => ~6.5us floor).
// ---------------------------------------------------------------------------

typedef _Float16 half_t;
typedef __attribute__((ext_vector_type(16))) _Float16 v16h;
typedef __attribute__((ext_vector_type(8)))  float    v8f;

#define Bq   48
#define Wd_  88
#define Cc   256
#define Hh   32
#define Ff   1024
#define Mtok (Bq * Wd_)     // 4224 tokens
#define RSQRT32 0.17677669529663687f

__device__ __forceinline__ float wred_sum(float v) {
#pragma unroll
  for (int m = 16; m >= 1; m >>= 1) v += __shfl_xor(v, m, 32);
  return v;
}
__device__ __forceinline__ float wred_max(float v) {
#pragma unroll
  for (int m = 16; m >= 1; m >>= 1) v = fmaxf(v, __shfl_xor(v, m, 32));
  return v;
}

// A fragment: 16x32 f16 tile from a row-major [*, lda] matrix.
// 16-bit A layout (ISA 7.12.2): lane m holds K 0-7 & 16-23 (lane m+16: +8),
// i.e. two contiguous 16-byte chunks -> two b128 loads.
__device__ __forceinline__ v16h load_a_frag(const half_t* A, int lda, int row0, int k0) {
  int lane = threadIdx.x & 31;
  int hl = lane >> 4, m = lane & 15;
  const half_t* base = A + (size_t)(row0 + m) * lda + k0 + hl * 8;
  union { v16h v; uint4 q[2]; } f;
  f.q[0] = *reinterpret_cast<const uint4*>(base);        // K 0..7   (+hl*8)
  f.q[1] = *reinterpret_cast<const uint4*>(base + 16);   // K 16..23 (+hl*8)
  return f.v;
}

// B fragment from pre-swizzled weights: lane's 8 dwords are contiguous.
// Bsw index (uint4 units): (((kt * ntiles + nt) * 32 + lane) * 2.
__device__ __forceinline__ v16h load_b_frag_swz(const uint4* Bsw, int ntiles, int kt, int nt) {
  int lane = threadIdx.x & 31;
  const uint4* base = Bsw + ((size_t)(kt * ntiles + nt) * 32 + lane) * 2;
  union { v16h v; uint4 q[2]; } f;
  f.q[0] = base[0];
  f.q[1] = base[1];
  return f.v;
}

// ---------------------------------------------------------------------------
// Swizzle f32 weights [K,N] into WMMA B-fragment order (f16 pairs per dword).
// B layout (matches A): lane n (0-15) holds K 0-7 & 16-23, lanes 16-31 K 8-15 & 24-31.
// out dword index i = ((kt*ntiles + nt)*32 + lane)*8 + j.
// ---------------------------------------------------------------------------
__global__ void swz_b_kernel(const float* __restrict__ W, unsigned int* __restrict__ out,
                             int K, int N) {
  int i = blockIdx.x * 256 + threadIdx.x;
  int total = (K * N) >> 1;
  if (i >= total) return;
  int j    = i & 7;
  int lane = (i >> 3) & 31;
  int blk  = i >> 8;
  int ntiles = N >> 4;
  int nt = blk % ntiles;
  int kt = blk / ntiles;
  int k   = kt * 32 + ((lane >> 4) << 3) + ((j >> 2) << 4) + 2 * (j & 3);
  int col = nt * 16 + (lane & 15);
  union { half_t h[2]; unsigned int u; } pk;
  pk.h[0] = (half_t)W[(size_t)k * N + col];
  pk.h[1] = (half_t)W[(size_t)(k + 1) * N + col];
  out[i] = pk.u;
}

// ---------------------------------------------------------------------------
// Row LayerNorm: one wave per row of 256; optional positional-embedding add
// (pe row = row % peW). Output f16.
// ---------------------------------------------------------------------------
__global__ void ln_rows_kernel(const float* __restrict__ X, const float* __restrict__ pe, int peW,
                               const float* __restrict__ g, const float* __restrict__ bb,
                               half_t* __restrict__ Y, int rows) {
  int row = blockIdx.x * 8 + (threadIdx.x >> 5);
  if (row >= rows) return;
  int lane = threadIdx.x & 31;
  const float* xr = X + (size_t)row * Cc;
  const float* pr = pe ? pe + (size_t)(row % peW) * Cc : nullptr;
  float v[8];
  float s = 0.f;
#pragma unroll
  for (int i = 0; i < 8; ++i) {
    int c = lane + 32 * i;
    v[i] = xr[c] + (pr ? pr[c] : 0.f);
    s += v[i];
  }
  float mu = wred_sum(s) * (1.f / Cc);
  float s2 = 0.f;
#pragma unroll
  for (int i = 0; i < 8; ++i) { float d = v[i] - mu; s2 += d * d; }
  float inv = rsqrtf(wred_sum(s2) * (1.f / Cc) + 1e-5f);
#pragma unroll
  for (int i = 0; i < 8; ++i) {
    int c = lane + 32 * i;
    Y[(size_t)row * Cc + c] = (half_t)((v[i] - mu) * inv * g[c] + bb[c]);
  }
}

// ---------------------------------------------------------------------------
// Generic WMMA GEMM: out = [residual +] act(A_f16[M,K] @ B[K,N] + bias)
// One wave computes a 32x64 strip (2 M-tiles x 4 N-tiles): 8 WMMAs per K-step
// against 2 A-frag + 4 B-frag b128-pair loads. act: 0 none, 1 exact GELU.
// Job guard is wave-uniform so EXEC is all-ones at every WMMA.
// ---------------------------------------------------------------------------
__global__ void gemm_wmma_kernel(const half_t* __restrict__ A, const uint4* __restrict__ Bsw,
                                 const float* __restrict__ bias, const float* __restrict__ resid,
                                 float* __restrict__ outF, half_t* __restrict__ outH,
                                 int M, int N, int K, int act) {
  int lane = threadIdx.x & 31;
  int wave = threadIdx.x >> 5;
  int mstrips = M >> 5;   // M multiple of 32
  int njobs   = N >> 6;   // N multiple of 64
  int job = blockIdx.x * 8 + wave;
  if (job >= mstrips * njobs) return;
  int mt = job / njobs, nj = job - mt * njobs;
  int row0 = mt << 5, col0 = nj << 6;
  int ntiles = N >> 4;
  int nt0 = nj << 2;

  v8f acc[2][4];
#pragma unroll
  for (int h = 0; h < 2; ++h)
#pragma unroll
    for (int t = 0; t < 4; ++t)
      acc[h][t] = (v8f){0.f, 0.f, 0.f, 0.f, 0.f, 0.f, 0.f, 0.f};

  for (int k0 = 0; k0 < K; k0 += 32) {
    int kt = k0 >> 5;
    v16h a0 = load_a_frag(A, K, row0, k0);
    v16h a1 = load_a_frag(A, K, row0 + 16, k0);
#pragma unroll
    for (int t = 0; t < 4; ++t) {
      v16h b = load_b_frag_swz(Bsw, ntiles, kt, nt0 + t);
      acc[0][t] = __builtin_amdgcn_wmma_f32_16x16x32_f16(false, a0, false, b,
                                                         (short)0, acc[0][t], false, false);
      acc[1][t] = __builtin_amdgcn_wmma_f32_16x16x32_f16(false, a1, false, b,
                                                         (short)0, acc[1][t], false, false);
    }
  }

  int hl = lane >> 4, nl = lane & 15;
#pragma unroll
  for (int h = 0; h < 2; ++h) {
#pragma unroll
    for (int t = 0; t < 4; ++t) {
      int col = col0 + t * 16 + nl;
      float bcol = bias ? bias[col] : 0.f;
#pragma unroll
      for (int r = 0; r < 8; ++r) {
        int row = row0 + h * 16 + r + hl * 8;   // C/D layout: VGPR r -> M=r / r+8
        float x = acc[h][t][r] + bcol;
        if (act) x = 0.5f * x * (1.f + erff(x * 0.70710678118654752f));
        if (resid) x += resid[(size_t)row * N + col];
        if (outF) outF[(size_t)row * N + col] = x;
        if (outH) outH[(size_t)row * N + col] = (half_t)x;
      }
    }
  }
}

// ---------------------------------------------------------------------------
// Fused cross attention: one block (8 waves) per (b,w) column.
// Stage 1: gather kv[h][c]=features[b,c,h,w], two LayerNorms -> knL/vnL (LDS f16)
// Stage 2: k = knL@Wk, v = vnL@Wv + bv via WMMA (A: ds_load_b128, B: swizzled L2)
// Stage 3: per-head (wave=head, lane=key) softmax attention over H=32 keys.
// 64KB static LDS; kn/vn/k/v never touch HBM.
// ---------------------------------------------------------------------------
__global__ void cross_attn_fused_kernel(const float* __restrict__ feat,
                                        const float* __restrict__ cpe,
                                        const float* __restrict__ g_kv, const float* __restrict__ b_kv,
                                        const float* __restrict__ g_vc, const float* __restrict__ b_vc,
                                        const uint4* __restrict__ WkS, const uint4* __restrict__ WvS,
                                        const float* __restrict__ bvc,
                                        const float* __restrict__ qv,
                                        half_t* __restrict__ co) {
  __shared__ half_t knL[Hh][Cc];
  __shared__ half_t vnL[Hh][Cc];
  __shared__ half_t kL[Hh][Cc];
  __shared__ half_t vL[Hh][Cc];

  int bw = blockIdx.x;
  int b = bw / Wd_, w = bw - b * Wd_;
  int lane = threadIdx.x & 31, wave = threadIdx.x >> 5;
  const float* fb = feat + ((size_t)b * Cc * Hh) * Wd_ + w;

  // ---- Stage 1: LN rows (wave per h-row, lane handles 8 channels) ----
  for (int h = wave; h < Hh; h += 8) {
    float v[8];
    float s = 0.f;
#pragma unroll
    for (int i = 0; i < 8; ++i) {
      int c = lane + 32 * i;
      v[i] = fb[((size_t)c * Hh + h) * Wd_];
      s += v[i];
    }
    float mu = wred_sum(s) * (1.f / Cc);
    float s2 = 0.f;
#pragma unroll
    for (int i = 0; i < 8; ++i) { float d = v[i] - mu; s2 += d * d; }
    float inv = rsqrtf(wred_sum(s2) * (1.f / Cc) + 1e-5f);
#pragma unroll
    for (int i = 0; i < 8; ++i) {
      int c = lane + 32 * i;
      vnL[h][c] = (half_t)((v[i] - mu) * inv * g_vc[c] + b_vc[c]);
    }
    float p[8];
    s = 0.f;
#pragma unroll
    for (int i = 0; i < 8; ++i) {
      int c = lane + 32 * i;
      p[i] = v[i] + cpe[h * Cc + c];
      s += p[i];
    }
    mu = wred_sum(s) * (1.f / Cc);
    s2 = 0.f;
#pragma unroll
    for (int i = 0; i < 8; ++i) { float d = p[i] - mu; s2 += d * d; }
    inv = rsqrtf(wred_sum(s2) * (1.f / Cc) + 1e-5f);
#pragma unroll
    for (int i = 0; i < 8; ++i) {
      int c = lane + 32 * i;
      knL[h][c] = (half_t)((p[i] - mu) * inv * g_kv[c] + b_kv[c]);
    }
  }
  __syncthreads();

  // ---- Stage 2: two 32x256x256 WMMA GEMMs (64 16x16 tiles over 8 waves) ----
  for (int t = wave; t < 64; t += 8) {
    int which = t >> 5;                   // 0: k = kn@Wk, 1: v = vn@Wv + bv
    int tt = t & 31, mt = tt & 1, nt = tt >> 1;
    const half_t* Am = which ? &vnL[0][0] : &knL[0][0];
    const uint4* Bm = which ? WvS : WkS;
    v8f acc = (v8f){0.f, 0.f, 0.f, 0.f, 0.f, 0.f, 0.f, 0.f};
    for (int k0 = 0; k0 < Cc; k0 += 32) {
      v16h a  = load_a_frag(Am, Cc, mt * 16, k0);
      v16h bf = load_b_frag_swz(Bm, Cc >> 4, k0 >> 5, nt);
      acc = __builtin_amdgcn_wmma_f32_16x16x32_f16(false, a, false, bf,
                                                   (short)0, acc, false, false);
    }
    int hl = lane >> 4, nl = lane & 15;
    int col = nt * 16 + nl;
    float badd = which ? bvc[col] : 0.f;
    half_t* dst = which ? &vL[0][0] : &kL[0][0];
#pragma unroll
    for (int r = 0; r < 8; ++r)
      dst[(size_t)(mt * 16 + r + hl * 8) * Cc + col] = (half_t)(acc[r] + badd);
  }
  __syncthreads();

  // ---- Stage 3: attention, wave = head, lane = key (then lane = dim) ----
  int head = wave;
  const float* qrow = qv + (size_t)bw * Cc + head * 32;
  float sc = 0.f;
#pragma unroll
  for (int d = 0; d < 32; ++d) sc += qrow[d] * (float)kL[lane][head * 32 + d];
  sc *= RSQRT32;
  float mx = wred_max(sc);
  float e = expf(sc - mx);
  float at = e / wred_sum(e);
  float o = 0.f;
#pragma unroll
  for (int kk = 0; kk < 32; ++kk)
    o += __shfl(at, kk, 32) * (float)vL[kk][head * 32 + lane];
  co[(size_t)bw * Cc + head * 32 + lane] = (half_t)o;
}

// ---------------------------------------------------------------------------
// Self attention over the 88 width tokens: one block per (b, head).
// Tiny (0.4 GF total) -> plain VALU with shfl softmax broadcast.
// ---------------------------------------------------------------------------
__global__ void self_attn_kernel(const half_t* __restrict__ qs, const half_t* __restrict__ ks,
                                 const half_t* __restrict__ vs, half_t* __restrict__ so) {
  __shared__ half_t qL[Wd_ * 32];
  __shared__ half_t kL[Wd_ * 32];
  __shared__ half_t vL[Wd_ * 32];
  int b = blockIdx.x >> 3, head = blockIdx.x & 7;
  int tid = threadIdx.x;
  for (int i = tid; i < Wd_ * 32; i += 256) {
    int tok = i >> 5, d = i & 31;
    size_t src = ((size_t)(b * Wd_ + tok)) * Cc + head * 32 + d;
    qL[i] = qs[src];
    kL[i] = ks[src];
    vL[i] = vs[src];
  }
  __syncthreads();
  int lane = tid & 31, wave = tid >> 5;
  for (int n = wave; n < Wd_; n += 8) {       // 11 rows per wave, uniform
    float s0, s1, s2 = -1e30f;
    {
      float a0 = 0.f, a1 = 0.f, a2 = 0.f;
      for (int d = 0; d < 32; ++d) {
        float q = (float)qL[n * 32 + d];
        a0 += q * (float)kL[(lane) * 32 + d];
        a1 += q * (float)kL[(lane + 32) * 32 + d];
        if (lane + 64 < Wd_) a2 += q * (float)kL[(lane + 64) * 32 + d];
      }
      s0 = a0 * RSQRT32;
      s1 = a1 * RSQRT32;
      if (lane + 64 < Wd_) s2 = a2 * RSQRT32;
    }
    float mx = wred_max(fmaxf(s0, fmaxf(s1, s2)));
    float e0 = expf(s0 - mx);
    float e1 = expf(s1 - mx);
    float e2 = (lane + 64 < Wd_) ? expf(s2 - mx) : 0.f;
    float inv = 1.f / wred_sum(e0 + e1 + e2);
    float a0 = e0 * inv, a1 = e1 * inv, a2 = e2 * inv;
    float o = 0.f;
    for (int m = 0; m < 32; ++m) o += __shfl(a0, m, 32) * (float)vL[m * 32 + lane];
    for (int m = 0; m < 32; ++m) o += __shfl(a1, m, 32) * (float)vL[(m + 32) * 32 + lane];
    for (int m = 0; m < Wd_ - 64; ++m) o += __shfl(a2, m, 32) * (float)vL[(m + 64) * 32 + lane];
    so[((size_t)(b * Wd_ + n)) * Cc + head * 32 + lane] = (half_t)o;
  }
}

// ---------------------------------------------------------------------------
extern "C" void kernel_launch(void* const* d_in, const int* in_sizes, int n_in,
                              void* d_out, int out_size, void* d_ws, size_t ws_size,
                              hipStream_t stream) {
  const float* pooled   = (const float*)d_in[0];
  const float* features = (const float*)d_in[1];
  const float* self_pe  = (const float*)d_in[2];
  const float* cross_pe = (const float*)d_in[3];
  const float* ln_qs_g  = (const float*)d_in[4];
  const float* ln_qs_b  = (const float*)d_in[5];
  const float* ln_vs_g  = (const float*)d_in[6];
  const float* ln_vs_b  = (const float*)d_in[7];
  const float* ln_qc_g  = (const float*)d_in[8];
  const float* ln_qc_b  = (const float*)d_in[9];
  const float* ln_kv_g  = (const float*)d_in[10];
  const float* ln_kv_b  = (const float*)d_in[11];
  const float* ln_vc_g  = (const float*)d_in[12];
  const float* ln_vc_b  = (const float*)d_in[13];
  const float* ln_ffn_g = (const float*)d_in[14];
  const float* ln_ffn_b = (const float*)d_in[15];
  const float* Wq_s = (const float*)d_in[16];
  const float* Wk_s = (const float*)d_in[17];
  const float* Wv_s = (const float*)d_in[18];
  const float* Wp_s = (const float*)d_in[19];
  const float* Wq_c = (const float*)d_in[20];
  const float* Wk_c = (const float*)d_in[21];
  const float* Wv_c = (const float*)d_in[22];
  const float* Wp_c = (const float*)d_in[23];
  const float* bq_s = (const float*)d_in[24];
  const float* bv_s = (const float*)d_in[25];
  const float* bp_s = (const float*)d_in[26];
  const float* bq_c = (const float*)d_in[27];
  const float* bv_c = (const float*)d_in[28];
  const float* bp_c = (const float*)d_in[29];
  const float* W1 = (const float*)d_in[30];
  const float* b1 = (const float*)d_in[31];
  const float* W2 = (const float*)d_in[32];
  const float* b2 = (const float*)d_in[33];

  // ---- workspace carve (~41MB) ----
  char* p = (char*)d_ws;
  auto carve = [&](size_t bytes) -> char* {
    char* r = p;
    p += (bytes + 255) & ~(size_t)255;
    return r;
  };
  const size_t CC = (size_t)Cc * Cc;
  uint4* wqsS = (uint4*)carve(CC * 2);
  uint4* wksS = (uint4*)carve(CC * 2);
  uint4* wvsS = (uint4*)carve(CC * 2);
  uint4* wpsS = (uint4*)carve(CC * 2);
  uint4* wqcS = (uint4*)carve(CC * 2);
  uint4* wkcS = (uint4*)carve(CC * 2);
  uint4* wvcS = (uint4*)carve(CC * 2);
  uint4* wpcS = (uint4*)carve(CC * 2);
  uint4* w1S  = (uint4*)carve((size_t)Cc * Ff * 2);
  uint4* w2S  = (uint4*)carve((size_t)Ff * Cc * 2);
  half_t* qnH  = (half_t*)carve((size_t)Mtok * Cc * 2);
  float*  qF   = (float*) carve((size_t)Mtok * Cc * 4);
  half_t* coH  = (half_t*)carve((size_t)Mtok * Cc * 2);
  float*  xF   = (float*) carve((size_t)Mtok * Cc * 4);
  half_t* qnsH = (half_t*)carve((size_t)Mtok * Cc * 2);
  half_t* vnsH = (half_t*)carve((size_t)Mtok * Cc * 2);
  half_t* qsH  = (half_t*)carve((size_t)Mtok * Cc * 2);
  half_t* ksH  = (half_t*)carve((size_t)Mtok * Cc * 2);
  half_t* vsH  = (half_t*)carve((size_t)Mtok * Cc * 2);
  half_t* soH  = (half_t*)carve((size_t)Mtok * Cc * 2);
  float*  x2F  = (float*) carve((size_t)Mtok * Cc * 4);
  half_t* hnH  = (half_t*)carve((size_t)Mtok * Cc * 2);
  half_t* h1H  = (half_t*)carve((size_t)Mtok * Ff * 2);

  auto swz = [&](const float* src, uint4* dst, int K, int N) {
    int ndw = (K * N) >> 1;
    swz_b_kernel<<<(ndw + 255) / 256, 256, 0, stream>>>(src, (unsigned int*)dst, K, N);
  };
  auto gemm = [&](const half_t* A, const uint4* Bw, const float* bias, const float* res,
                  float* oF, half_t* oH, int M, int N, int K, int act) {
    int jobs = (M / 32) * (N / 64);
    gemm_wmma_kernel<<<(jobs + 7) / 8, 256, 0, stream>>>(A, Bw, bias, res, oF, oH, M, N, K, act);
  };
  const int lnBlocks = (Mtok + 7) / 8;

  // ---- weights -> swizzled f16 fragments ----
  swz(Wq_s, wqsS, Cc, Cc); swz(Wk_s, wksS, Cc, Cc);
  swz(Wv_s, wvsS, Cc, Cc); swz(Wp_s, wpsS, Cc, Cc);
  swz(Wq_c, wqcS, Cc, Cc); swz(Wk_c, wkcS, Cc, Cc);
  swz(Wv_c, wvcS, Cc, Cc); swz(Wp_c, wpcS, Cc, Cc);
  swz(W1, w1S, Cc, Ff);    swz(W2, w2S, Ff, Cc);

  // ---- cross attention ----
  ln_rows_kernel<<<lnBlocks, 256, 0, stream>>>(pooled, nullptr, 1, ln_qc_g, ln_qc_b, qnH, Mtok);
  gemm(qnH, wqcS, bq_c, nullptr, qF, nullptr, Mtok, Cc, Cc, 0);            // q = qn@Wq_c + bq_c
  cross_attn_fused_kernel<<<Mtok, 256, 0, stream>>>(features, cross_pe,
                                                    ln_kv_g, ln_kv_b, ln_vc_g, ln_vc_b,
                                                    wkcS, wvcS, bv_c, qF, coH);
  gemm(coH, wpcS, bp_c, pooled, xF, nullptr, Mtok, Cc, Cc, 0);             // x = co@Wp_c+bp_c+pooled

  // ---- self attention ----
  ln_rows_kernel<<<lnBlocks, 256, 0, stream>>>(xF, self_pe, Wd_, ln_qs_g, ln_qs_b, qnsH, Mtok);
  ln_rows_kernel<<<lnBlocks, 256, 0, stream>>>(xF, nullptr, 1, ln_vs_g, ln_vs_b, vnsH, Mtok);
  gemm(qnsH, wqsS, bq_s, nullptr, nullptr, qsH, Mtok, Cc, Cc, 0);
  gemm(qnsH, wksS, nullptr, nullptr, nullptr, ksH, Mtok, Cc, Cc, 0);
  gemm(vnsH, wvsS, bv_s, nullptr, nullptr, vsH, Mtok, Cc, Cc, 0);
  self_attn_kernel<<<Bq * 8, 256, 0, stream>>>(qsH, ksH, vsH, soH);
  gemm(soH, wpsS, bp_s, xF, x2F, nullptr, Mtok, Cc, Cc, 0);                // x2 = so@Wp_s+bp_s+x

  // ---- FFN ----
  ln_rows_kernel<<<lnBlocks, 256, 0, stream>>>(x2F, nullptr, 1, ln_ffn_g, ln_ffn_b, hnH, Mtok);
  gemm(hnH, w1S, b1, nullptr, nullptr, h1H, Mtok, Ff, Cc, 1);              // gelu(hn@W1+b1)
  gemm(h1H, w2S, b2, x2F, (float*)d_out, nullptr, Mtok, Cc, Ff, 0);        // out = .@W2+b2+x2
}
